// Attention_88098369175915
// MI455X (gfx1250) — compile-verified, matrix operand order checked by reading, not run on hardware
//
#include <hip/hip_runtime.h>
#include <hip/hip_bf16.h>

#define BB 16
#define SS 626
#define SP 640         // S padded to 32 for V^T layout
#define DD 768
#define HH 12
#define HD 64
#define ST 40          // ceil(S/16) q/k tiles
#define MT 626         // (B*S)/16 exactly (10016/16)
#define COEFF_MAX 0.25f

typedef __attribute__((ext_vector_type(16))) _Float16 v16h;
typedef __attribute__((ext_vector_type(8)))  float    v8f;

#if defined(__gfx1250__) && __has_builtin(__builtin_amdgcn_tensor_load_to_lds)
#define HAVE_TDM 1
#else
#define HAVE_TDM 0
#endif

#if HAVE_TDM
typedef __attribute__((ext_vector_type(4))) unsigned int u32x4;
typedef __attribute__((ext_vector_type(8))) int i32x8;
typedef __attribute__((ext_vector_type(4))) int i32x4;
#endif

__device__ __forceinline__ int laneid() { return (int)(threadIdx.x & 31u); }

__device__ __forceinline__ v8f wmma_f16(v16h a, v16h b, v8f c) {
  return __builtin_amdgcn_wmma_f32_16x16x32_f16(
      false, a, false, b, (short)0, c, false, false);
}

// Stage a 16-row x DD-col f16 slab from global into LDS via the Tensor Data Mover.
__device__ __forceinline__ void tdm_stage_tile16(const _Float16* gsrc, _Float16* lds_dst) {
#if HAVE_TDM
  unsigned long long ga = (unsigned long long)(uintptr_t)gsrc;
  unsigned lo = (unsigned)(uintptr_t)lds_dst;
  u32x4 g0;
  g0[0] = 1u;                                     // count=1, user mode
  g0[1] = lo;                                     // lds_addr
  g0[2] = (unsigned)(ga & 0xFFFFFFFFu);           // global_addr[31:0]
  g0[3] = (unsigned)((ga >> 32) & 0x1FFFFFFu) | (2u << 30);  // global_addr[56:32], type=2
  i32x8 g1;
  g1[0] = (int)(1u << 16);                        // data_size = 1 -> 2 bytes
  g1[1] = (int)((unsigned)DD << 16);              // tensor_dim0[15:0]
  g1[2] = (int)(16u << 16);                       // tensor_dim1[15:0] = 16 rows
  g1[3] = (int)((unsigned)DD << 16);              // tile_dim0 = DD
  g1[4] = 16;                                     // tile_dim1 = 16
  g1[5] = DD;                                     // tensor_dim0_stride[31:0]
  g1[6] = 0;
  g1[7] = 0;
  i32x4 z4 = {0, 0, 0, 0};
#if __clang_major__ >= 23
  i32x8 z8 = {0, 0, 0, 0, 0, 0, 0, 0};
  __builtin_amdgcn_tensor_load_to_lds(g0, g1, z4, z4, z8, 0);
#else
  __builtin_amdgcn_tensor_load_to_lds(g0, g1, z4, z4, 0);
#endif
  __builtin_amdgcn_s_wait_tensorcnt(0);
#else
  (void)gsrc; (void)lds_dst;
#endif
}

// ---------------- f32 -> f16 convert ----------------
__global__ void cvt_f32_f16(const float* __restrict__ src, _Float16* __restrict__ dst, int n) {
  int i = blockIdx.x * blockDim.x + threadIdx.x;
  int stride = gridDim.x * blockDim.x;
  for (; i < n; i += stride) dst[i] = (_Float16)src[i];
}

// ---------------- f32 -> f16 transpose (W[k][n] -> WT[n][k]) ----------------
__global__ void cvt_transpose(const float* __restrict__ W, _Float16* __restrict__ WT) {
  __shared__ float tile[16][17];
  int kt = blockIdx.x, ntb = blockIdx.y;
  int tx = threadIdx.x & 15, ty = threadIdx.x >> 4;
  tile[ty][tx] = W[(size_t)(kt * 16 + ty) * DD + ntb * 16 + tx];
  __syncthreads();
  WT[(size_t)(ntb * 16 + ty) * DD + kt * 16 + tx] = (_Float16)tile[tx][ty];
}

// ---------------- GEMM (f16 in, f16 out, f32 bias), B pre-transposed ----------------
// grid(MT), block 384 (12 waves); each wave owns 4 N-tiles -> 4 WMMAs per A-fragment.
// vt_out != 0: scatter result into V^T layout [b][d][s] with stride SP.
__global__ void gemm_f16out(const _Float16* __restrict__ X, const _Float16* __restrict__ WT,
                            const float* __restrict__ bias, _Float16* __restrict__ Y, int vt_out) {
  __shared__ _Float16 Atile[16 * DD];
  int mt = blockIdx.x;
  int nt0 = (threadIdx.x >> 5) * 4;
  const _Float16* Abase = X + (size_t)mt * 16 * DD;
#if HAVE_TDM
  if (threadIdx.x < 32) tdm_stage_tile16(Abase, Atile);
#else
  for (int i = threadIdx.x * 8; i < 16 * DD; i += 384 * 8)
    *reinterpret_cast<uint4*>(&Atile[i]) = *reinterpret_cast<const uint4*>(&Abase[i]);
#endif
  __syncthreads();

  int lane = laneid();
  int r = lane & 15, ko = (lane >> 4) * 8;
  const _Float16* B0 = WT + (size_t)((nt0 + 0) * 16 + r) * DD;
  const _Float16* B1 = WT + (size_t)((nt0 + 1) * 16 + r) * DD;
  const _Float16* B2 = WT + (size_t)((nt0 + 2) * 16 + r) * DD;
  const _Float16* B3 = WT + (size_t)((nt0 + 3) * 16 + r) * DD;
  v8f acc[4] = {{}, {}, {}, {}};
  for (int kk = 0; kk < DD; kk += 32) {
    __builtin_prefetch(B0 + kk + 128, 0, 1);
    v16h a, b0, b1, b2, b3;
#pragma unroll
    for (int i = 0; i < 8; ++i) a[i]     = Atile[r * DD + kk + ko + i];
#pragma unroll
    for (int i = 0; i < 8; ++i) a[8 + i] = Atile[r * DD + kk + 16 + ko + i];
#pragma unroll
    for (int i = 0; i < 8; ++i) { b0[i] = B0[kk + ko + i]; b0[8 + i] = B0[kk + 16 + ko + i]; }
#pragma unroll
    for (int i = 0; i < 8; ++i) { b1[i] = B1[kk + ko + i]; b1[8 + i] = B1[kk + 16 + ko + i]; }
#pragma unroll
    for (int i = 0; i < 8; ++i) { b2[i] = B2[kk + ko + i]; b2[8 + i] = B2[kk + 16 + ko + i]; }
#pragma unroll
    for (int i = 0; i < 8; ++i) { b3[i] = B3[kk + ko + i]; b3[8 + i] = B3[kk + 16 + ko + i]; }
    acc[0] = wmma_f16(a, b0, acc[0]);
    acc[1] = wmma_f16(a, b1, acc[1]);
    acc[2] = wmma_f16(a, b2, acc[2]);
    acc[3] = wmma_f16(a, b3, acc[3]);
  }
  int col = lane & 15, rb = (lane >> 4) * 8;
#pragma unroll
  for (int n = 0; n < 4; ++n) {
    int nc = (nt0 + n) * 16 + col;
    float bvv = bias[nc];
#pragma unroll
    for (int v = 0; v < 8; ++v) {
      int gr = mt * 16 + rb + v;
      float val = acc[n][v] + bvv;
      if (vt_out) {
        int bidx = gr / SS, s = gr - bidx * SS;
        Y[((size_t)bidx * DD + nc) * SP + s] = (_Float16)val;
      } else {
        Y[(size_t)gr * DD + nc] = (_Float16)val;
      }
    }
  }
}

// ---------------- output projection GEMM -> f32 ----------------
__global__ void gemm_f32out(const _Float16* __restrict__ X, const _Float16* __restrict__ WT,
                            const float* __restrict__ bias, float* __restrict__ out) {
  __shared__ _Float16 Atile[16 * DD];
  int mt = blockIdx.x;
  int nt0 = (threadIdx.x >> 5) * 4;
  const _Float16* Abase = X + (size_t)mt * 16 * DD;
#if HAVE_TDM
  if (threadIdx.x < 32) tdm_stage_tile16(Abase, Atile);
#else
  for (int i = threadIdx.x * 8; i < 16 * DD; i += 384 * 8)
    *reinterpret_cast<uint4*>(&Atile[i]) = *reinterpret_cast<const uint4*>(&Abase[i]);
#endif
  __syncthreads();

  int lane = laneid();
  int r = lane & 15, ko = (lane >> 4) * 8;
  const _Float16* B0 = WT + (size_t)((nt0 + 0) * 16 + r) * DD;
  const _Float16* B1 = WT + (size_t)((nt0 + 1) * 16 + r) * DD;
  const _Float16* B2 = WT + (size_t)((nt0 + 2) * 16 + r) * DD;
  const _Float16* B3 = WT + (size_t)((nt0 + 3) * 16 + r) * DD;
  v8f acc[4] = {{}, {}, {}, {}};
  for (int kk = 0; kk < DD; kk += 32) {
    __builtin_prefetch(B0 + kk + 128, 0, 1);
    v16h a, b0, b1, b2, b3;
#pragma unroll
    for (int i = 0; i < 8; ++i) a[i]     = Atile[r * DD + kk + ko + i];
#pragma unroll
    for (int i = 0; i < 8; ++i) a[8 + i] = Atile[r * DD + kk + 16 + ko + i];
#pragma unroll
    for (int i = 0; i < 8; ++i) { b0[i] = B0[kk + ko + i]; b0[8 + i] = B0[kk + 16 + ko + i]; }
#pragma unroll
    for (int i = 0; i < 8; ++i) { b1[i] = B1[kk + ko + i]; b1[8 + i] = B1[kk + 16 + ko + i]; }
#pragma unroll
    for (int i = 0; i < 8; ++i) { b2[i] = B2[kk + ko + i]; b2[8 + i] = B2[kk + 16 + ko + i]; }
#pragma unroll
    for (int i = 0; i < 8; ++i) { b3[i] = B3[kk + ko + i]; b3[8 + i] = B3[kk + 16 + ko + i]; }
    acc[0] = wmma_f16(a, b0, acc[0]);
    acc[1] = wmma_f16(a, b1, acc[1]);
    acc[2] = wmma_f16(a, b2, acc[2]);
    acc[3] = wmma_f16(a, b3, acc[3]);
  }
  int col = lane & 15, rb = (lane >> 4) * 8;
#pragma unroll
  for (int n = 0; n < 4; ++n) {
    int nc = (nt0 + n) * 16 + col;
    float bvv = bias[nc];
#pragma unroll
    for (int v = 0; v < 8; ++v)
      out[(size_t)(mt * 16 + rb + v) * DD + nc] = acc[n][v] + bvv;
  }
}

// ---------------- raw scores = Q K^T / 8 -> probs region ----------------
// grid(ST, H, B), block 256 = 8 waves; Q fragments hoisted out of the key-tile loop
__global__ void scores_kernel(const _Float16* __restrict__ Q, const _Float16* __restrict__ K,
                              float* __restrict__ probs) {
  int qt = blockIdx.x, h = blockIdx.y, b = blockIdx.z;
  int wave = threadIdx.x >> 5;
  int q0 = qt * 16;
  int qcl = SS - 1 - q0; if (qcl > 15) qcl = 15;
  int lane = laneid();
  int col = lane & 15, rb = (lane >> 4) * 8, ko = (lane >> 4) * 8;
  int qr = lane & 15; if (qr > qcl) qr = qcl;
  const _Float16* pa = Q + ((size_t)(b * SS + q0) + qr) * DD + h * HD;
  v16h a0, a1;
#pragma unroll
  for (int i = 0; i < 8; ++i) { a0[i] = pa[ko + i];      a0[8 + i] = pa[16 + ko + i]; }
#pragma unroll
  for (int i = 0; i < 8; ++i) { a1[i] = pa[32 + ko + i]; a1[8 + i] = pa[48 + ko + i]; }
  for (int kt = wave; kt < ST; kt += 8) {
    int k0 = kt * 16;
    int kcl = SS - 1 - k0; if (kcl > 15) kcl = 15;
    int kr = lane & 15; if (kr > kcl) kr = kcl;
    const _Float16* pb = K + ((size_t)(b * SS + k0) + kr) * DD + h * HD;
    v16h b0, b1;
#pragma unroll
    for (int i = 0; i < 8; ++i) { b0[i] = pb[ko + i];      b0[8 + i] = pb[16 + ko + i]; }
#pragma unroll
    for (int i = 0; i < 8; ++i) { b1[i] = pb[32 + ko + i]; b1[8 + i] = pb[48 + ko + i]; }
    v8f acc = {};
    acc = wmma_f16(a0, b0, acc);
    acc = wmma_f16(a1, b1, acc);
#pragma unroll
    for (int v = 0; v < 8; ++v) {
      int qg = q0 + rb + v, kc = k0 + col;
      if (qg < SS && kc < SS)
        probs[(((size_t)b * HH + h) * SS + qg) * SS + kc] = acc[v] * 0.125f;
    }
  }
}

// ---------------- fused row stats + head-0 adjust + softmax normalize ----------------
__global__ void stats_norm_kernel(const int* __restrict__ mask, float* __restrict__ probs,
                                  float* __restrict__ col0) {
  int q = blockIdx.x, h = blockIdx.y, b = blockIdx.z;
  float* row = probs + (((size_t)b * HH + h) * SS + q) * SS;
  __shared__ float red[256];
  int t = threadIdx.x;
  float x0 = (t       < SS) ? row[t]       : -1e30f;
  float x1 = (t + 256 < SS) ? row[t + 256] : -1e30f;
  float x2 = (t + 512 < SS) ? row[t + 512] : -1e30f;
  float m = fmaxf(x0, fmaxf(x1, x2));
  red[t] = m; __syncthreads();
  for (int s = 128; s > 0; s >>= 1) { if (t < s) red[t] = fmaxf(red[t], red[t + s]); __syncthreads(); }
  m = red[0]; __syncthreads();
  if (h == 0) {
    const int* mrow = mask + b * (SS - 1);
    float add = m * COEFF_MAX;
    if (t < SS       && (t == 0 || mrow[t - 1] == 0)) x0 += add;
    if (t + 256 < SS && mrow[t + 255] == 0)           x1 += add;
    if (t + 512 < SS && mrow[t + 511] == 0)           x2 += add;
    float m2 = fmaxf(x0, fmaxf(x1, x2));
    red[t] = m2; __syncthreads();
    for (int s = 128; s > 0; s >>= 1) { if (t < s) red[t] = fmaxf(red[t], red[t + s]); __syncthreads(); }
    m = red[0]; __syncthreads();
  }
  float e0 = (t       < SS) ? __expf(x0 - m) : 0.f;
  float e1 = (t + 256 < SS) ? __expf(x1 - m) : 0.f;
  float e2 = (t + 512 < SS) ? __expf(x2 - m) : 0.f;
  red[t] = e0 + e1 + e2; __syncthreads();
  for (int s = 128; s > 0; s >>= 1) { if (t < s) red[t] += red[t + s]; __syncthreads(); }
  float inv = 1.0f / red[0];
  if (t == 0) col0[((size_t)b * HH + h) * SS + q] = x0;
  if (t       < SS) row[t]       = e0 * inv;
  if (t + 256 < SS) row[t + 256] = e1 * inv;
  if (t + 512 < SS) row[t + 512] = e2 * inv;
}

// ---------------- ctx = probs @ V ----------------
// grid(ST, H, B), block 32 (one wave): wave owns all 4 hd-tiles -> probs A-frag loaded once
__global__ void ctx_kernel(const float* __restrict__ probs, const _Float16* __restrict__ Vt,
                           _Float16* __restrict__ ctx) {
  int qt = blockIdx.x, h = blockIdx.y, b = blockIdx.z;
  int q0 = qt * 16;
  int lane = laneid();
  int r = lane & 15, ko = (lane >> 4) * 8;
  int rq = q0 + r; if (rq >= SS) rq = SS - 1;
  const float* prow = probs + (((size_t)b * HH + h) * SS + rq) * SS;
  const _Float16* v0 = Vt + ((size_t)b * DD + h * HD + r) * SP;
  const _Float16* v1 = v0 + (size_t)16 * SP;
  const _Float16* v2 = v0 + (size_t)32 * SP;
  const _Float16* v3 = v0 + (size_t)48 * SP;
  v8f acc[4] = {{}, {}, {}, {}};
  for (int k0 = 0; k0 < SS; k0 += 32) {
    __builtin_prefetch(prow + k0 + 64, 0, 1);
    v16h a, b0, b1, b2, b3;
#pragma unroll
    for (int i = 0; i < 8; ++i) { int k = k0 + ko + i;      a[i]     = (k < SS) ? (_Float16)prow[k] : (_Float16)0.f; }
#pragma unroll
    for (int i = 0; i < 8; ++i) { int k = k0 + 16 + ko + i; a[8 + i] = (k < SS) ? (_Float16)prow[k] : (_Float16)0.f; }
#pragma unroll
    for (int i = 0; i < 8; ++i) { b0[i] = v0[k0 + ko + i]; b0[8 + i] = v0[k0 + 16 + ko + i]; }
#pragma unroll
    for (int i = 0; i < 8; ++i) { b1[i] = v1[k0 + ko + i]; b1[8 + i] = v1[k0 + 16 + ko + i]; }
#pragma unroll
    for (int i = 0; i < 8; ++i) { b2[i] = v2[k0 + ko + i]; b2[8 + i] = v2[k0 + 16 + ko + i]; }
#pragma unroll
    for (int i = 0; i < 8; ++i) { b3[i] = v3[k0 + ko + i]; b3[8 + i] = v3[k0 + 16 + ko + i]; }
    acc[0] = wmma_f16(a, b0, acc[0]);
    acc[1] = wmma_f16(a, b1, acc[1]);
    acc[2] = wmma_f16(a, b2, acc[2]);
    acc[3] = wmma_f16(a, b3, acc[3]);
  }
  int col = lane & 15, rb = (lane >> 4) * 8;
#pragma unroll
  for (int n = 0; n < 4; ++n) {
#pragma unroll
    for (int v = 0; v < 8; ++v) {
      int qr = q0 + rb + v;
      if (qr < SS)
        ctx[((size_t)b * SS + qr) * DD + h * HD + n * 16 + col] = (_Float16)acc[n][v];
    }
  }
}

// ---------------- contribution = softmax over q of col0 ----------------
__global__ void contrib_kernel(const float* __restrict__ col0, float* __restrict__ contrib) {
  int h = blockIdx.x, b = blockIdx.y;
  const float* src = col0 + ((size_t)b * HH + h) * SS;
  float* dst = contrib + ((size_t)b * HH + h) * SS;
  __shared__ float red[256];
  int t = threadIdx.x;
  float m = -1e30f;
  for (int q = t; q < SS; q += 256) m = fmaxf(m, src[q]);
  red[t] = m; __syncthreads();
  for (int s = 128; s > 0; s >>= 1) { if (t < s) red[t] = fmaxf(red[t], red[t + s]); __syncthreads(); }
  m = red[0]; __syncthreads();
  float sum = 0.f;
  for (int q = t; q < SS; q += 256) sum += __expf(src[q] - m);
  red[t] = sum; __syncthreads();
  for (int s = 128; s > 0; s >>= 1) { if (t < s) red[t] += red[t + s]; __syncthreads(); }
  float inv = 1.0f / red[0];
  for (int q = t; q < SS; q += 256) dst[q] = __expf(src[q] - m) * inv;
}

extern "C" void kernel_launch(void* const* d_in, const int* in_sizes, int n_in,
                              void* d_out, int out_size, void* d_ws, size_t ws_size,
                              hipStream_t stream) {
  const float* hs = (const float*)d_in[0];
  const int*  mask = (const int*)d_in[1];
  const float* Wq = (const float*)d_in[2];
  const float* bq = (const float*)d_in[3];
  const float* Wk = (const float*)d_in[4];
  const float* bk = (const float*)d_in[5];
  const float* Wv = (const float*)d_in[6];
  const float* bv = (const float*)d_in[7];
  const float* Wo = (const float*)d_in[8];
  const float* bo = (const float*)d_in[9];

  const size_t nHS = (size_t)BB * SS * DD;
  const size_t nW  = (size_t)DD * DD;
  const size_t nVt = (size_t)BB * DD * SP;
  const size_t nRows = (size_t)BB * HH * SS;

  float* out_attn = (float*)d_out;
  float* probs    = out_attn + nHS;
  float* contrib  = probs + (size_t)BB * HH * SS * SS;

  char* w = (char*)d_ws;
  _Float16* hsf = (_Float16*)w; w += nHS * 2;
  _Float16* wqt = (_Float16*)w; w += nW * 2;
  _Float16* wkt = (_Float16*)w; w += nW * 2;
  _Float16* wvt = (_Float16*)w; w += nW * 2;
  _Float16* wot = (_Float16*)w; w += nW * 2;
  _Float16* Qf  = (_Float16*)w; w += nHS * 2;
  _Float16* Kf  = (_Float16*)w; w += nHS * 2;
  _Float16* Vt  = (_Float16*)w; w += nVt * 2;
  _Float16* Cf  = (_Float16*)w; w += nHS * 2;
  float* col0   = (float*)w;    w += nRows * 4;

  // 1) precision conversion + weight transposes (contiguous WMMA operands)
  cvt_f32_f16<<<2048, 256, 0, stream>>>(hs, hsf, (int)nHS);
  cvt_transpose<<<dim3(48, 48), 256, 0, stream>>>(Wq, wqt);
  cvt_transpose<<<dim3(48, 48), 256, 0, stream>>>(Wk, wkt);
  cvt_transpose<<<dim3(48, 48), 256, 0, stream>>>(Wv, wvt);
  cvt_transpose<<<dim3(48, 48), 256, 0, stream>>>(Wo, wot);

  // 2) QKV projections (WMMA, TDM-staged A slab, 4 N-tiles/wave); V emitted transposed
  gemm_f16out<<<dim3(MT), 384, 0, stream>>>(hsf, wqt, bq, Qf, 0);
  gemm_f16out<<<dim3(MT), 384, 0, stream>>>(hsf, wkt, bk, Kf, 0);
  gemm_f16out<<<dim3(MT), 384, 0, stream>>>(hsf, wvt, bv, Vt, 1);

  // 3) raw scores into probs region (WMMA, hoisted Q fragments)
  scores_kernel<<<dim3(ST, HH, BB), 256, 0, stream>>>(Qf, Kf, probs);

  // 4) fused stats + head-0 adjustment + softmax normalize + col0 capture
  stats_norm_kernel<<<dim3(SS, HH, BB), 256, 0, stream>>>(mask, probs, col0);

  // 5) ctx = probs @ V (WMMA, 4 hd-tiles/wave)
  ctx_kernel<<<dim3(ST, HH, BB), 32, 0, stream>>>(probs, Vt, Cf);

  // 6) output projection (WMMA)
  gemm_f32out<<<dim3(MT), 384, 0, stream>>>(Cf, wot, bo, out_attn);

  // 7) contribution = column softmax of scores col 0
  contrib_kernel<<<dim3(HH, BB), 256, 0, stream>>>(col0, contrib);
}
